// LinearLoopLayer_21251498180715
// MI455X (gfx1250) — matrix operand order repaired and checked by reading, not compiled, not run on hardware
//
#include <hip/hip_runtime.h>

#define B_DIM   4096
#define IN_DIM  2048
#define OUT_DIM 2048

typedef __attribute__((ext_vector_type(16))) __bf16 v16bf;
typedef __attribute__((ext_vector_type(8)))  __bf16 v8bf;
typedef __attribute__((ext_vector_type(4)))  __bf16 v4bf;
typedef __attribute__((ext_vector_type(8)))  float  v8f;

#define BM 256            // batch rows per block
#define BN 128            // out cols per block
#define BK 32             // K per step (one bf16 WMMA)
#define MT 4              // 16-row tiles per wave  (wave tile 64x64)
#define NT 4              // 16-col tiles per wave
#define LDS_STRIDE 40     // 32 bf16 + 8 pad (80B row stride -> conflict-free ds_load_b128)

// A-fragment (16x32 bf16, MxK), documented wave32 layout:
// lanes 0-15:  row=lane,    halfs[0..7]=K0..7,  halfs[8..15]=K16..23
// lanes 16-31: row=lane-16, halfs[0..7]=K8..15, halfs[8..15]=K24..31
__device__ __forceinline__ v16bf load_a_frag(const __bf16* base, int row, int lhalf) {
  v8bf lo = *reinterpret_cast<const v8bf*>(base + row * LDS_STRIDE + lhalf * 8);
  v8bf hi = *reinterpret_cast<const v8bf*>(base + row * LDS_STRIDE + 16 + lhalf * 8);
  v16bf r;
#pragma unroll
  for (int i = 0; i < 8; ++i) { r[i] = lo[i]; r[8 + i] = hi[i]; }
  return r;
}

// B-fragment (32x16 bf16, KxN): lanes 0-15 hold K=0..7,16..23 of col=lane,
// lanes 16-31 hold K=8..15,24..31 of col=lane-16 (same K order as A).
__device__ __forceinline__ v16bf load_b_frag(const __bf16* base, int row, int lhalf) {
  v8bf lo = *reinterpret_cast<const v8bf*>(base + row * LDS_STRIDE + lhalf * 8);
  v8bf hi = *reinterpret_cast<const v8bf*>(base + row * LDS_STRIDE + 16 + lhalf * 8);
  v16bf r;
#pragma unroll
  for (int i = 0; i < 8; ++i) { r[i] = lo[i]; r[8 + i] = hi[i]; }
  return r;
}

__global__ __launch_bounds__(256)
void linear_bias_bf16x3_kernel(const float* __restrict__ X,
                               const float* __restrict__ W,
                               const float* __restrict__ bias,
                               float* __restrict__ Y)
{
  __shared__ __bf16 sXh[BM * LDS_STRIDE];   // 20 KB
  __shared__ __bf16 sXl[BM * LDS_STRIDE];   // 20 KB
  __shared__ __bf16 sWh[BN * LDS_STRIDE];   // 10 KB
  __shared__ __bf16 sWl[BN * LDS_STRIDE];   // 10 KB

  const int tid   = threadIdx.x;
  const int lane  = tid & 31;
  const int wave  = tid >> 5;
  const int waveM = wave & 3;   // 4 waves along M -> 64 rows each
  const int waveN = wave >> 2;  // 2 waves along N -> 64 cols each
  const int lhalf = lane >> 4;  // 0/1 lane half
  const int lrow  = lane & 15;

  const int gRow0 = blockIdx.y * BM;  // batch tile
  const int gCol0 = blockIdx.x * BN;  // OUT tile

  const float4* Xv = reinterpret_cast<const float4*>(X);
  const float4* Wv = reinterpret_cast<const float4*>(W);
  const int ld4 = IN_DIM / 4;

  // Global staging: X tile = 256 rows x 8 float4 = 2048 -> 8/thread;
  //                 W tile = 128 rows x 8 float4 = 1024 -> 4/thread.
  float4 xr[8], wr[4];
#pragma unroll
  for (int j = 0; j < 8; ++j) {
    int f = tid + 256 * j;
    int r = f >> 3, c4 = f & 7;
    xr[j] = Xv[(size_t)(gRow0 + r) * ld4 + c4];
  }
#pragma unroll
  for (int j = 0; j < 4; ++j) {
    int f = tid + 256 * j;
    int r = f >> 3, c4 = f & 7;
    wr[j] = Wv[(size_t)(gCol0 + r) * ld4 + c4];
  }

  v8f acc[MT][NT];
#pragma unroll
  for (int mt = 0; mt < MT; ++mt)
#pragma unroll
    for (int nt = 0; nt < NT; ++nt)
      acc[mt][nt] = (v8f){0.f, 0.f, 0.f, 0.f, 0.f, 0.f, 0.f, 0.f};

  const int ksteps = IN_DIM / BK;  // 64
  for (int kk = 0; kk < ksteps; ++kk) {
    // Split staged fp32 into (hi, lo) bf16 pairs and store to LDS (ds_store_b64).
#pragma unroll
    for (int j = 0; j < 8; ++j) {
      int f = tid + 256 * j;
      int r = f >> 3, c = (f & 7) * 4;
      float xs[4] = {xr[j].x, xr[j].y, xr[j].z, xr[j].w};
      v4bf xh, xl;
#pragma unroll
      for (int e = 0; e < 4; ++e) {
        __bf16 h = (__bf16)xs[e];
        xh[e] = h;
        xl[e] = (__bf16)(xs[e] - (float)h);
      }
      *reinterpret_cast<v4bf*>(&sXh[r * LDS_STRIDE + c]) = xh;
      *reinterpret_cast<v4bf*>(&sXl[r * LDS_STRIDE + c]) = xl;
    }
#pragma unroll
    for (int j = 0; j < 4; ++j) {
      int f = tid + 256 * j;
      int r = f >> 3, c = (f & 7) * 4;
      float ws[4] = {wr[j].x, wr[j].y, wr[j].z, wr[j].w};
      v4bf wh, wl;
#pragma unroll
      for (int e = 0; e < 4; ++e) {
        __bf16 g = (__bf16)ws[e];
        wh[e] = g;
        wl[e] = (__bf16)(ws[e] - (float)g);
      }
      *reinterpret_cast<v4bf*>(&sWh[r * LDS_STRIDE + c]) = wh;
      *reinterpret_cast<v4bf*>(&sWl[r * LDS_STRIDE + c]) = wl;
    }
    __syncthreads();

    // Prefetch next K-step from global; latency hidden under the 48-WMMA loop.
    if (kk + 1 < ksteps) {
      int kb4 = (kk + 1) * (BK / 4);
#pragma unroll
      for (int j = 0; j < 8; ++j) {
        int f = tid + 256 * j;
        int r = f >> 3, c4 = f & 7;
        xr[j] = Xv[(size_t)(gRow0 + r) * ld4 + kb4 + c4];
      }
#pragma unroll
      for (int j = 0; j < 4; ++j) {
        int f = tid + 256 * j;
        int r = f >> 3, c4 = f & 7;
        wr[j] = Wv[(size_t)(gCol0 + r) * ld4 + kb4 + c4];
      }
    }

    // A fragments for this wave's four 16-row tiles (hi and lo split).
    v16bf ah[MT], al[MT];
#pragma unroll
    for (int mt = 0; mt < MT; ++mt) {
      int row = waveM * 64 + mt * 16 + lrow;
      ah[mt] = load_a_frag(sXh, row, lhalf);
      al[mt] = load_a_frag(sXl, row, lhalf);
    }

    // 48 WMMAs per wave per K-step (3 split terms x 4x4 tiles);
    // mt innermost so consecutive WMMAs hit different accumulators.
#pragma unroll
    for (int nt = 0; nt < NT; ++nt) {
      int col = waveN * 64 + nt * 16 + lrow;
      v16bf bh = load_b_frag(sWh, col, lhalf);
      v16bf bl = load_b_frag(sWl, col, lhalf);
#pragma unroll
      for (int mt = 0; mt < MT; ++mt)
        acc[mt][nt] = __builtin_amdgcn_wmma_f32_16x16x32_bf16(
            false, ah[mt], false, bh, (short)0, acc[mt][nt], false, false);
#pragma unroll
      for (int mt = 0; mt < MT; ++mt)
        acc[mt][nt] = __builtin_amdgcn_wmma_f32_16x16x32_bf16(
            false, ah[mt], false, bl, (short)0, acc[mt][nt], false, false);
#pragma unroll
      for (int mt = 0; mt < MT; ++mt)
        acc[mt][nt] = __builtin_amdgcn_wmma_f32_16x16x32_bf16(
            false, al[mt], false, bh, (short)0, acc[mt][nt], false, false);
    }
    __syncthreads();
  }

  // Epilogue: C/D layout is VGPR r -> M = r (lanes 0-15) / r+8 (lanes 16-31), N = lane&15.
#pragma unroll
  for (int nt = 0; nt < NT; ++nt) {
    int col = gCol0 + waveN * 64 + nt * 16 + lrow;
    float bv = bias[col];
#pragma unroll
    for (int mt = 0; mt < MT; ++mt) {
      int rbase = gRow0 + waveM * 64 + mt * 16 + lhalf * 8;
#pragma unroll
      for (int r = 0; r < 8; ++r) {
        Y[(size_t)(rbase + r) * OUT_DIM + col] = acc[mt][nt][r] + bv;
      }
    }
  }
}

extern "C" void kernel_launch(void* const* d_in, const int* in_sizes, int n_in,
                              void* d_out, int out_size, void* d_ws, size_t ws_size,
                              hipStream_t stream) {
  const float* x = (const float*)d_in[0];
  const float* w = (const float*)d_in[1];
  const float* b = (const float*)d_in[2];
  float* y = (float*)d_out;
  dim3 grid(OUT_DIM / BN, B_DIM / BM);  // (16, 16)
  linear_bias_bf16x3_kernel<<<grid, 256, 0, stream>>>(x, w, b, y);
  (void)in_sizes; (void)n_in; (void)out_size; (void)d_ws; (void)ws_size;
}
